// DecoderLayer_53841710023063
// MI455X (gfx1250) — matrix operand order repaired and checked by reading, not compiled, not run on hardware
//
#include <hip/hip_runtime.h>
#include <hip/hip_bf16.h>
#include <math.h>

typedef __bf16 bf16_t;
typedef __attribute__((ext_vector_type(16))) __bf16 v16bf;
typedef __attribute__((ext_vector_type(8)))  float  v8f;

#define D_MODEL 512
#define D_FF    2048
#define N_HEAD  8
#define D_HEAD  64
#define BATCH   4
#define L_Q     1024
#define L_M     4096
#define LN_EPS  1e-3f
#define ATTN_SCALE 0.04419417382415922f  /* 1/sqrt(512): ref divides by sqrt(D) */

// Load a 32-byte WMMA fragment from two 16B-aligned LDS runs.
__device__ __forceinline__ v16bf ld_frag(const bf16_t* p0, const bf16_t* p1) {
  union { v16bf v; uint4 q[2]; } u;
  u.q[0] = *(const uint4*)p0;
  u.q[1] = *(const uint4*)p1;
  return u.v;
}

// ---------------------------------------------------------------------------
// elementwise helpers
// ---------------------------------------------------------------------------
__global__ void k_cvt_bf16(const float* __restrict__ a, bf16_t* __restrict__ o, int n) {
  int i = blockIdx.x * blockDim.x + threadIdx.x;
  if (i < n) o[i] = (bf16_t)a[i];
}

__global__ void k_add_bf16(const float* __restrict__ a, const float* __restrict__ b,
                           bf16_t* __restrict__ o, int n) {
  int i = blockIdx.x * blockDim.x + threadIdx.x;
  if (i < n) o[i] = (bf16_t)(a[i] + b[i]);
}

// ---------------------------------------------------------------------------
// GEMM: C[M,N] = A[M,K](bf16,row-major) * W[K,N](bf16,row-major) + bias[N]
// block = 128 threads (4 waves), block tile 128x128, K-step 32.
// wave w computes rows [w*32, w*32+32) x all 128 cols (2x8 WMMA accumulators)
// -> 16 WMMAs per K-step vs 4 ds_load_b128 + 128 b16 B loads.
// ---------------------------------------------------------------------------
#define GT_K 32
#define BM 128
#define BN 128
#define A_STRIDE 40    // 32 + 8 pad (80B rows, 16B aligned)
#define B_STRIDE 136   // 128 + 8 pad (272B rows, 16B aligned)

template<bool RELU, bool OUTF, bool OUTB>
__global__ void __launch_bounds__(128)
k_gemm(const bf16_t* __restrict__ A, const bf16_t* __restrict__ W,
       const float* __restrict__ bias, float* __restrict__ Cf,
       bf16_t* __restrict__ Cb, int M, int N, int K) {
  __shared__ alignas(16) bf16_t As[BM * A_STRIDE];
  __shared__ alignas(16) bf16_t Bs[GT_K * B_STRIDE];
  const int t = threadIdx.x;
  const int w = t >> 5, lane = t & 31;
  const int m0 = blockIdx.y * BM, n0 = blockIdx.x * BN;
  const int arow = lane & 15;
  const int khi  = (lane >> 4) << 3;   // 0 or 8: A-frag k-offset by lane-half
  const int rowoff = (lane >> 4) << 3; // C layout: reg r -> row r + rowoff

  v8f acc[2][8];
  #pragma unroll
  for (int mf = 0; mf < 2; mf++)
    #pragma unroll
    for (int nf = 0; nf < 8; nf++)
      #pragma unroll
      for (int e = 0; e < 8; e++) acc[mf][nf][e] = 0.f;

  for (int k0 = 0; k0 < K; k0 += GT_K) {
    __syncthreads();
    { // stage A tile 128x32: thread t stages one full row
      const bf16_t* src = A + (size_t)(m0 + t) * K + k0;
      bf16_t* dst = As + t * A_STRIDE;
      #pragma unroll
      for (int i = 0; i < 4; i++)
        *(uint4*)(dst + 8 * i) = *(const uint4*)(src + 8 * i);
    }
    { // stage B tile 32x128: thread t stages 4 chunks of 8
      int r = t >> 2, cb = (t & 3) * 32;
      const bf16_t* src = W + (size_t)(k0 + r) * N + n0 + cb;
      bf16_t* dst = Bs + r * B_STRIDE + cb;
      #pragma unroll
      for (int i = 0; i < 4; i++)
        *(uint4*)(dst + 8 * i) = *(const uint4*)(src + 8 * i);
    }
    __syncthreads();

    v16bf af[2];
    #pragma unroll
    for (int mf = 0; mf < 2; mf++) {
      const bf16_t* ap = As + (w * 32 + mf * 16 + arow) * A_STRIDE;
      af[mf] = ld_frag(ap + khi, ap + 16 + khi);
    }
    #pragma unroll
    for (int nf = 0; nf < 8; nf++) {
      const bf16_t* bp = Bs + nf * 16 + arow;  // lane%16 = N, elements = K (table order)
      v16bf bfr;
      #pragma unroll
      for (int s2 = 0; s2 < 16; s2++) bfr[s2] = bp[((s2 & 8) * 2 + khi + (s2 & 7)) * B_STRIDE];
      #pragma unroll
      for (int mf = 0; mf < 2; mf++)
        acc[mf][nf] = __builtin_amdgcn_wmma_f32_16x16x32_bf16(
            false, af[mf], false, bfr, (short)0, acc[mf][nf], false, false);
    }
  }

  #pragma unroll
  for (int mf = 0; mf < 2; mf++)
    #pragma unroll
    for (int nf = 0; nf < 8; nf++)
      #pragma unroll
      for (int r = 0; r < 8; r++) {
        int gm = m0 + w * 32 + mf * 16 + r + rowoff;
        int gn = n0 + nf * 16 + (lane & 15);
        float v = acc[mf][nf][r] + bias[gn];
        if (RELU) v = v > 0.f ? v : 0.f;
        size_t idx = (size_t)gm * N + gn;
        if (OUTF) Cf[idx] = v;
        if (OUTB) Cb[idx] = (bf16_t)v;
      }
}

// ---------------------------------------------------------------------------
// Flash attention: softmax(Q K^T * scale) V, per (batch, head).
// Q/K/V/O laid out [B*L, D_MODEL] with head h at columns [h*64, h*64+64).
// block = 128 threads (4 waves); block handles 128 q rows; wave owns 32 rows
// (2 q-fragments), so each staged K/V tile and B-fragment feeds 2 WMMAs.
// K is staged TRANSPOSED in LDS (KsT[dh][key]) ; score B-fragments are
// 16 strided b16 loads (stride TS); V stays row-major, its B-frags also strided.
// ---------------------------------------------------------------------------
#define QROWS 128  // q rows per block
#define TS 72      // 64 + 8 pad

__global__ void __launch_bounds__(128)
k_attn(const bf16_t* __restrict__ Q, const bf16_t* __restrict__ Kg,
       const bf16_t* __restrict__ V, bf16_t* __restrict__ O,
       int Lq, int Lk, float scale) {
  __shared__ alignas(16) bf16_t Qs[QROWS * TS];
  __shared__ alignas(16) bf16_t KsT[64 * TS];       // [dh][key]
  __shared__ alignas(16) bf16_t Vs[64 * TS];        // [key][dh]
  __shared__ alignas(16) bf16_t Ps[4 * 32 * TS];    // per-wave P (32 rows)
  const int t = threadIdx.x, w = t >> 5, lane = t & 31;
  const int q0 = blockIdx.x * QROWS;
  const int h  = blockIdx.y;
  const int b  = blockIdx.z;
  const size_t qbase = ((size_t)b * Lq + q0) * D_MODEL + h * D_HEAD;
  const size_t kbase = ((size_t)b * Lk) * D_MODEL + h * D_HEAD;
  const int arow = lane & 15;
  const int khi  = (lane >> 4) << 3;
  const int rowoff = (lane >> 4) << 3;

  { // stage Q tile 128x64 row-major: thread t stages row t
    const bf16_t* src = Q + qbase + (size_t)t * D_MODEL;
    bf16_t* dst = Qs + t * TS;
    #pragma unroll
    for (int i = 0; i < 8; i++) *(uint4*)(dst + 8 * i) = *(const uint4*)(src + 8 * i);
  }
  __syncthreads();

  // preload Q A-fragments: [mf][dh-step], contiguous 16B runs
  v16bf qa[2][2];
  #pragma unroll
  for (int mf = 0; mf < 2; mf++) {
    const bf16_t* qp = Qs + (w * 32 + mf * 16 + arow) * TS;
    #pragma unroll
    for (int st = 0; st < 2; st++)
      qa[mf][st] = ld_frag(qp + st * 32 + khi, qp + st * 32 + 16 + khi);
  }

  v8f o[2][4];
  float mst[2][8], lst[2][8];
  #pragma unroll
  for (int mf = 0; mf < 2; mf++) {
    #pragma unroll
    for (int d = 0; d < 4; d++)
      #pragma unroll
      for (int e = 0; e < 8; e++) o[mf][d][e] = 0.f;
    #pragma unroll
    for (int r = 0; r < 8; r++) { mst[mf][r] = -1e30f; lst[mf][r] = 0.f; }
  }

  for (int kt = 0; kt < Lk; kt += 64) {
    __syncthreads();  // prior PV reads of KsT/Vs done
    { // stage K (transposed) and V (row-major) tiles, 64 keys x 64 dh
      int r = t >> 1, cb = (t & 1) * 32;
      const bf16_t* ksrc = Kg + kbase + (size_t)(kt + r) * D_MODEL + cb;
      const bf16_t* vsrc = V  + kbase + (size_t)(kt + r) * D_MODEL + cb;
      bf16_t* vdst = Vs + r * TS + cb;
      #pragma unroll
      for (int i = 0; i < 4; i++) {
        union { uint4 q; bf16_t e[8]; } u;
        u.q = *(const uint4*)(ksrc + 8 * i);
        #pragma unroll
        for (int e2 = 0; e2 < 8; e2++)
          KsT[(cb + 8 * i + e2) * TS + r] = u.e[e2];   // [dh][key]
        *(uint4*)(vdst + 8 * i) = *(const uint4*)(vsrc + 8 * i);
      }
    }
    __syncthreads();

    // S = Q * K^T : 2 q-frags x 4 key-chunks, reduce over dh in 2 steps
    v8f s[2][4];
    #pragma unroll
    for (int c = 0; c < 4; c++) {
      #pragma unroll
      for (int mf = 0; mf < 2; mf++)
        #pragma unroll
        for (int e = 0; e < 8; e++) s[mf][c][e] = 0.f;
      #pragma unroll
      for (int st = 0; st < 2; st++) {
        // B frag: lane%16 = key, elements = dh (reduction, table order) from KsT
        const bf16_t* kp = KsT + (st * 32) * TS + c * 16 + arow;
        v16bf kb;
        #pragma unroll
        for (int s2 = 0; s2 < 16; s2++) kb[s2] = kp[((s2 & 8) * 2 + khi + (s2 & 7)) * TS];
        #pragma unroll
        for (int mf = 0; mf < 2; mf++)
          s[mf][c] = __builtin_amdgcn_wmma_f32_16x16x32_bf16(
              false, qa[mf][st], false, kb, (short)0, s[mf][c], false, false);
      }
    }

    // online softmax in C-fragment layout (reg r <-> row r+rowoff)
    #pragma unroll
    for (int mf = 0; mf < 2; mf++) {
      #pragma unroll
      for (int c = 0; c < 4; c++)
        #pragma unroll
        for (int e = 0; e < 8; e++) s[mf][c][e] *= scale;
      #pragma unroll
      for (int r = 0; r < 8; r++) {
        float mx = s[mf][0][r];
        #pragma unroll
        for (int c = 1; c < 4; c++) mx = fmaxf(mx, s[mf][c][r]);
        #pragma unroll
        for (int msk = 1; msk <= 8; msk <<= 1) mx = fmaxf(mx, __shfl_xor(mx, msk, 32));
        float mnew = fmaxf(mst[mf][r], mx);
        float alpha = __expf(mst[mf][r] - mnew);
        mst[mf][r] = mnew;
        float rsum = 0.f;
        #pragma unroll
        for (int c = 0; c < 4; c++) {
          float pv = __expf(s[mf][c][r] - mnew);
          s[mf][c][r] = pv;
          rsum += pv;
        }
        #pragma unroll
        for (int msk = 1; msk <= 8; msk <<= 1) rsum += __shfl_xor(rsum, msk, 32);
        lst[mf][r] = lst[mf][r] * alpha + rsum;
        #pragma unroll
        for (int d = 0; d < 4; d++) o[mf][d][r] *= alpha;
      }
    }

    // P: C layout -> LDS [row][key] (scalar b16 scatter, unavoidable)
    {
      bf16_t* pw = Ps + w * 32 * TS;
      #pragma unroll
      for (int mf = 0; mf < 2; mf++)
        #pragma unroll
        for (int c = 0; c < 4; c++)
          #pragma unroll
          for (int r = 0; r < 8; r++)
            pw[(mf * 16 + r + rowoff) * TS + c * 16 + (lane & 15)] =
                (bf16_t)s[mf][c][r];
    }
    __syncthreads();

    // O += P * V : each V B-frag feeds 2 WMMAs
    #pragma unroll
    for (int st = 0; st < 2; st++) {
      v16bf pa[2];
      #pragma unroll
      for (int mf = 0; mf < 2; mf++) {
        const bf16_t* pp = Ps + (w * 32 + mf * 16 + arow) * TS;
        pa[mf] = ld_frag(pp + st * 32 + khi, pp + st * 32 + 16 + khi);
      }
      #pragma unroll
      for (int d = 0; d < 4; d++) {
        // B frag: lane%16 = dh, elements = key (reduction, table order) from Vs
        const bf16_t* vp = Vs + (st * 32) * TS + d * 16 + arow;
        v16bf vb;
        #pragma unroll
        for (int s2 = 0; s2 < 16; s2++) vb[s2] = vp[((s2 & 8) * 2 + khi + (s2 & 7)) * TS];
        #pragma unroll
        for (int mf = 0; mf < 2; mf++)
          o[mf][d] = __builtin_amdgcn_wmma_f32_16x16x32_bf16(
              false, pa[mf], false, vb, (short)0, o[mf][d], false, false);
      }
    }
  }

  // normalize + store concatenated-head output (bf16, feeds Wo GEMM)
  #pragma unroll
  for (int mf = 0; mf < 2; mf++)
    #pragma unroll
    for (int d = 0; d < 4; d++)
      #pragma unroll
      for (int r = 0; r < 8; r++) {
        float v = o[mf][d][r] / lst[mf][r];
        size_t idx = qbase
                   + (size_t)(w * 32 + mf * 16 + r + rowoff) * D_MODEL
                   + d * 16 + (lane & 15);
        O[idx] = (bf16_t)v;
      }
}

// ---------------------------------------------------------------------------
// y = LayerNorm(x + r) * g + b ; one wave per row (D=512 -> 16 elems/lane)
// ---------------------------------------------------------------------------
template<bool OUTB>
__global__ void __launch_bounds__(256)
k_add_ln(const float* __restrict__ X, const float* __restrict__ R,
         const float* __restrict__ G, const float* __restrict__ Bv,
         float* __restrict__ Of, bf16_t* __restrict__ Ob, int rows) {
  int w = threadIdx.x >> 5, lane = threadIdx.x & 31;
  int row = blockIdx.x * 8 + w;
  if (row >= rows) return;
  const float* xp = X + (size_t)row * D_MODEL + lane * 16;
  const float* rp = R + (size_t)row * D_MODEL + lane * 16;
  float v[16];
  float s = 0.f;
  #pragma unroll
  for (int i = 0; i < 16; i++) { v[i] = xp[i] + rp[i]; s += v[i]; }
  #pragma unroll
  for (int msk = 1; msk <= 16; msk <<= 1) s += __shfl_xor(s, msk, 32);
  float mean = s * (1.f / (float)D_MODEL);
  float q = 0.f;
  #pragma unroll
  for (int i = 0; i < 16; i++) { float d = v[i] - mean; q += d * d; }
  #pragma unroll
  for (int msk = 1; msk <= 16; msk <<= 1) q += __shfl_xor(q, msk, 32);
  float rstd = rsqrtf(q * (1.f / (float)D_MODEL) + LN_EPS);
  #pragma unroll
  for (int i = 0; i < 16; i++) {
    int col = lane * 16 + i;
    float y = (v[i] - mean) * rstd * G[col] + Bv[col];
    size_t idx = (size_t)row * D_MODEL + col;
    Of[idx] = y;
    if (OUTB) Ob[idx] = (bf16_t)y;
  }
}

// ---------------------------------------------------------------------------
// host orchestration
// ---------------------------------------------------------------------------
extern "C" void kernel_launch(void* const* d_in, const int* in_sizes, int n_in,
                              void* d_out, int out_size, void* d_ws, size_t ws_size,
                              hipStream_t stream) {
  (void)in_sizes; (void)n_in; (void)out_size; (void)ws_size;
  const float* QO      = (const float*)d_in[0];
  const float* memory  = (const float*)d_in[1];
  const float* pos_enc = (const float*)d_in[2];
  const float* QO_enc  = (const float*)d_in[3];
  const float* sWq = (const float*)d_in[4];  const float* sbq = (const float*)d_in[5];
  const float* sWk = (const float*)d_in[6];  const float* sbk = (const float*)d_in[7];
  const float* sWv = (const float*)d_in[8];  const float* sbv = (const float*)d_in[9];
  const float* sWo = (const float*)d_in[10]; const float* sbo = (const float*)d_in[11];
  const float* cWq = (const float*)d_in[12]; const float* cbq = (const float*)d_in[13];
  const float* cWk = (const float*)d_in[14]; const float* cbk = (const float*)d_in[15];
  const float* cWv = (const float*)d_in[16]; const float* cbv = (const float*)d_in[17];
  const float* cWo = (const float*)d_in[18]; const float* cbo = (const float*)d_in[19];
  const float* fW1 = (const float*)d_in[20]; const float* fb1 = (const float*)d_in[21];
  const float* fW2 = (const float*)d_in[22]; const float* fb2 = (const float*)d_in[23];
  const float* g1 = (const float*)d_in[24];  const float* b1 = (const float*)d_in[25];
  const float* g2 = (const float*)d_in[26];  const float* b2 = (const float*)d_in[27];
  const float* g3 = (const float*)d_in[28];  const float* b3 = (const float*)d_in[29];
  float* out = (float*)d_out;

  const int BLQ = BATCH * L_Q;   // 4096 rows
  const int BLM = BATCH * L_M;   // 16384 rows

  char* p = (char*)d_ws;
  auto alloc = [&](size_t bytes) -> char* {
    char* r = p; p += (bytes + 255) & ~(size_t)255; return r;
  };
  // bf16 activations
  bf16_t* qk_b  = (bf16_t*)alloc((size_t)BLQ * D_MODEL * 2);
  bf16_t* qo_b  = (bf16_t*)alloc((size_t)BLQ * D_MODEL * 2);
  bf16_t* sq_b  = (bf16_t*)alloc((size_t)BLQ * D_MODEL * 2);
  bf16_t* sk_b  = (bf16_t*)alloc((size_t)BLQ * D_MODEL * 2);
  bf16_t* sv_b  = (bf16_t*)alloc((size_t)BLQ * D_MODEL * 2);
  bf16_t* sat_b = (bf16_t*)alloc((size_t)BLQ * D_MODEL * 2);
  bf16_t* qc_b  = (bf16_t*)alloc((size_t)BLQ * D_MODEL * 2);
  bf16_t* kc_b  = (bf16_t*)alloc((size_t)BLM * D_MODEL * 2);
  bf16_t* mem_b = (bf16_t*)alloc((size_t)BLM * D_MODEL * 2);
  bf16_t* cq_b  = (bf16_t*)alloc((size_t)BLQ * D_MODEL * 2);
  bf16_t* ck_b  = (bf16_t*)alloc((size_t)BLM * D_MODEL * 2);
  bf16_t* cv_b  = (bf16_t*)alloc((size_t)BLM * D_MODEL * 2);
  bf16_t* cat_b = (bf16_t*)alloc((size_t)BLQ * D_MODEL * 2);
  bf16_t* x2_b  = (bf16_t*)alloc((size_t)BLQ * D_MODEL * 2);
  bf16_t* h_b   = (bf16_t*)alloc((size_t)BLQ * D_FF * 2);
  // fp32 activations
  float* a1_f = (float*)alloc((size_t)BLQ * D_MODEL * 4);
  float* x1_f = (float*)alloc((size_t)BLQ * D_MODEL * 4);
  float* a2_f = (float*)alloc((size_t)BLQ * D_MODEL * 4);
  float* x2_f = (float*)alloc((size_t)BLQ * D_MODEL * 4);
  float* f_f  = (float*)alloc((size_t)BLQ * D_MODEL * 4);
  // bf16 weights
  bf16_t* wsq = (bf16_t*)alloc((size_t)D_MODEL * D_MODEL * 2);
  bf16_t* wsk = (bf16_t*)alloc((size_t)D_MODEL * D_MODEL * 2);
  bf16_t* wsv = (bf16_t*)alloc((size_t)D_MODEL * D_MODEL * 2);
  bf16_t* wso = (bf16_t*)alloc((size_t)D_MODEL * D_MODEL * 2);
  bf16_t* wcq = (bf16_t*)alloc((size_t)D_MODEL * D_MODEL * 2);
  bf16_t* wck = (bf16_t*)alloc((size_t)D_MODEL * D_MODEL * 2);
  bf16_t* wcv = (bf16_t*)alloc((size_t)D_MODEL * D_MODEL * 2);
  bf16_t* wco = (bf16_t*)alloc((size_t)D_MODEL * D_MODEL * 2);
  bf16_t* w1  = (bf16_t*)alloc((size_t)D_MODEL * D_FF * 2);
  bf16_t* w2  = (bf16_t*)alloc((size_t)D_FF * D_MODEL * 2);

  dim3 blk128(128);
  #define CVT(src, dst, n)    k_cvt_bf16<<<((n)+255)/256, 256, 0, stream>>>(src, dst, n)
  #define ADDB(a, bb, dst, n) k_add_bf16<<<((n)+255)/256, 256, 0, stream>>>(a, bb, dst, n)
  #define GEMM_B(A, W, bias, Cb, M, N, K) \
    k_gemm<false,false,true><<<dim3((N)/BN,(M)/BM), blk128, 0, stream>>>(A, W, bias, nullptr, Cb, M, N, K)
  #define GEMM_B_RELU(A, W, bias, Cb, M, N, K) \
    k_gemm<true,false,true><<<dim3((N)/BN,(M)/BM), blk128, 0, stream>>>(A, W, bias, nullptr, Cb, M, N, K)
  #define GEMM_F(A, W, bias, Cf, M, N, K) \
    k_gemm<false,true,false><<<dim3((N)/BN,(M)/BM), blk128, 0, stream>>>(A, W, bias, Cf, nullptr, M, N, K)

  // ---- weight conversions ----
  CVT(sWq, wsq, D_MODEL * D_MODEL); CVT(sWk, wsk, D_MODEL * D_MODEL);
  CVT(sWv, wsv, D_MODEL * D_MODEL); CVT(sWo, wso, D_MODEL * D_MODEL);
  CVT(cWq, wcq, D_MODEL * D_MODEL); CVT(cWk, wck, D_MODEL * D_MODEL);
  CVT(cWv, wcv, D_MODEL * D_MODEL); CVT(cWo, wco, D_MODEL * D_MODEL);
  CVT(fW1, w1, D_MODEL * D_FF);     CVT(fW2, w2, D_FF * D_MODEL);

  // ---- self-attention: Q=K=QO+QO_enc, V=QO ----
  ADDB(QO, QO_enc, qk_b, BLQ * D_MODEL);
  CVT(QO, qo_b, BLQ * D_MODEL);
  GEMM_B(qk_b, wsq, sbq, sq_b, BLQ, D_MODEL, D_MODEL);
  GEMM_B(qk_b, wsk, sbk, sk_b, BLQ, D_MODEL, D_MODEL);
  GEMM_B(qo_b, wsv, sbv, sv_b, BLQ, D_MODEL, D_MODEL);
  k_attn<<<dim3(L_Q / QROWS, N_HEAD, BATCH), blk128, 0, stream>>>(
      sq_b, sk_b, sv_b, sat_b, L_Q, L_Q, ATTN_SCALE);
  GEMM_F(sat_b, wso, sbo, a1_f, BLQ, D_MODEL, D_MODEL);
  k_add_ln<false><<<BLQ / 8, 256, 0, stream>>>(a1_f, QO, g1, b1, x1_f, nullptr, BLQ);

  // ---- cross-attention: Q=x1+QO_enc, K=memory+pos_enc, V=memory ----
  ADDB(x1_f, QO_enc, qc_b, BLQ * D_MODEL);
  ADDB(memory, pos_enc, kc_b, BLM * D_MODEL);
  CVT(memory, mem_b, BLM * D_MODEL);
  GEMM_B(qc_b, wcq, cbq, cq_b, BLQ, D_MODEL, D_MODEL);
  GEMM_B(kc_b, wck, cbk, ck_b, BLM, D_MODEL, D_MODEL);
  GEMM_B(mem_b, wcv, cbv, cv_b, BLM, D_MODEL, D_MODEL);
  k_attn<<<dim3(L_Q / QROWS, N_HEAD, BATCH), blk128, 0, stream>>>(
      cq_b, ck_b, cv_b, cat_b, L_Q, L_M, ATTN_SCALE);
  GEMM_F(cat_b, wco, cbo, a2_f, BLQ, D_MODEL, D_MODEL);
  k_add_ln<true><<<BLQ / 8, 256, 0, stream>>>(a2_f, x1_f, g2, b2, x2_f, x2_b, BLQ);

  // ---- FFN + final add&norm ----
  GEMM_B_RELU(x2_b, w1, fb1, h_b, BLQ, D_FF, D_MODEL);
  GEMM_F(h_b, w2, fb2, f_f, BLQ, D_MODEL, D_FF);
  k_add_ln<false><<<BLQ / 8, 256, 0, stream>>>(f_f, x2_f, g3, b3, out, nullptr, BLQ);

  #undef CVT
  #undef ADDB
  #undef GEMM_B
  #undef GEMM_B_RELU
  #undef GEMM_F
}
